// MultiHead_Forget_Attn_1554778161394
// MI455X (gfx1250) — matrix operand order repaired and checked by reading, not compiled, not run on hardware
//
#include <hip/hip_runtime.h>

// ---------------------------------------------------------------------------
// MultiHead "forget" attention (retention decay), MI455X / gfx1250, wave32.
// All matmuls: v_wmma_f32_16x16x32_bf16 (f32 accumulate); GEMMs and attn@V register-blocked
// 4 N-tiles per A-fragment. Scores never touch HBM: a 16xS stripe lives in
// LDS through both softmaxes + cumsum; Q stripe staged to LDS via the Tensor
// Data Mover (tensor_load_to_lds + s_wait_tensorcnt). The attention matrix
// (a reference output) is written exactly once and re-read once for attn@V.
// B=4, S=1024, d=1024, H=16, dk=64.
// ---------------------------------------------------------------------------

typedef __attribute__((ext_vector_type(16))) __bf16 v16bf;
typedef __attribute__((ext_vector_type(8)))  __bf16 v8bf;
typedef __attribute__((ext_vector_type(4)))  __bf16 v4bf;
typedef __attribute__((ext_vector_type(8)))  float  v8f;
typedef __attribute__((ext_vector_type(4)))  float  v4f;
typedef __attribute__((ext_vector_type(4)))  unsigned int v4u;
typedef __attribute__((ext_vector_type(8)))  int    v8i;
typedef __attribute__((ext_vector_type(4)))  int    v4i;

__device__ __forceinline__ __bf16 f2bf(float f) {
  unsigned u = __builtin_bit_cast(unsigned, f);
  unsigned r = (u + 0x7FFFu + ((u >> 16) & 1u)) >> 16;   // round-to-nearest-even
  return __builtin_bit_cast(__bf16, (unsigned short)r);
}

// A-fragment (16x32 bf16, M = lane&15): element j -> K = half*16+j.
// Two contiguous 16B loads from a row-major [M,K] row.
__device__ __forceinline__ v16bf load_a_frag(const __bf16* row, int k0, int hf) {
  v8bf lo = *(const v8bf*)(row + k0 + hf * 16);
  v8bf hi = *(const v8bf*)(row + k0 + hf * 16 + 8);
  return __builtin_shufflevector(lo, hi, 0,1,2,3,4,5,6,7,8,9,10,11,12,13,14,15);
}

__device__ __forceinline__ v8f wmma_bf16(v16bf a, v16bf b, v8f c) {
  return __builtin_amdgcn_wmma_f32_16x16x32_bf16(false, a, false, b,
                                                 (short)0, c, false, false);
}

__device__ __forceinline__ float wave_max(float x) {
  #pragma unroll
  for (int o = 16; o > 0; o >>= 1) x = fmaxf(x, __shfl_xor(x, o, 32));
  return x;
}
__device__ __forceinline__ float wave_sum(float x) {
  #pragma unroll
  for (int o = 16; o > 0; o >>= 1) x += __shfl_xor(x, o, 32);
  return x;
}

// ---------------------------------------------------------------------------
// f32 -> bf16 elementwise convert (x4 vectorized)
// ---------------------------------------------------------------------------
__global__ void f32_to_bf16_kernel(const float* __restrict__ in,
                                   __bf16* __restrict__ out, int n) {
  int i = (blockIdx.x * blockDim.x + threadIdx.x) * 4;
  if (i + 3 < n) {
    v4f f = *(const v4f*)(in + i);
    v4bf o;
    #pragma unroll
    for (int j = 0; j < 4; ++j) o[j] = f2bf(f[j]);
    *(v4bf*)(out + i) = o;
  } else {
    for (; i < n; ++i) out[i] = f2bf(in[i]);
  }
}

// ---------------------------------------------------------------------------
// GEMM: out[m,n] = sum_k A[m,k] * W[n,k] + bias[n]     (X @ W^T + b)
// A: [Mdim,Kdim] bf16 row-major; W: [Ndim,Kdim] bf16 row-major.
// Register blocking: each wave owns 16x64 (one A-frag feeds 4 WMMAs).
// Block = 4 waves = 16x256 of output. K-loop prefetches 4 tiles ahead with
// high-locality hint (WGP-scope prefetch pulls into all cache levels).
// out_mode 0: bf16 at [B,H,S,dk] (head-split Q/K/V);  1: f32 [Mdim,Ndim].
// ---------------------------------------------------------------------------
__global__ void __launch_bounds__(128)
gemm_bf16_kernel(const __bf16* __restrict__ A, const __bf16* __restrict__ W,
                 const float* __restrict__ bias, void* __restrict__ out,
                 int Mdim, int Ndim, int Kdim, int out_mode,
                 int S, int H, int dk) {
  int lane = threadIdx.x & 31;
  int wave = threadIdx.x >> 5;
  int r = lane & 15, hf = lane >> 4;
  int m0 = blockIdx.x << 4;
  int n0 = (blockIdx.y << 8) + (wave << 6);

  const __bf16* arow = A + (size_t)(m0 + r) * Kdim;
  const __bf16* br0  = W + (size_t)(n0 +  0 + r) * Kdim;
  const __bf16* br1  = W + (size_t)(n0 + 16 + r) * Kdim;
  const __bf16* br2  = W + (size_t)(n0 + 32 + r) * Kdim;
  const __bf16* br3  = W + (size_t)(n0 + 48 + r) * Kdim;

  v8f acc0 = {}, acc1 = {}, acc2 = {}, acc3 = {};
  for (int k0 = 0; k0 < Kdim; k0 += 32) {
    if (k0 + 128 < Kdim) {                       // gfx1250 global_prefetch_b8
      __builtin_prefetch((const void*)(arow + k0 + 128), 0, 3);
      __builtin_prefetch((const void*)(br0  + k0 + 128), 0, 3);
      __builtin_prefetch((const void*)(br2  + k0 + 128), 0, 3);
    }
    v16bf a = load_a_frag(arow, k0, hf);
    int kb = k0 + hf * 16;                       // B-frag: 16 contiguous K
    acc0 = wmma_bf16(a, *(const v16bf*)(br0 + kb), acc0);
    acc1 = wmma_bf16(a, *(const v16bf*)(br1 + kb), acc1);
    acc2 = wmma_bf16(a, *(const v16bf*)(br2 + kb), acc2);
    acc3 = wmma_bf16(a, *(const v16bf*)(br3 + kb), acc3);
  }

  v8f accs[4] = {acc0, acc1, acc2, acc3};
  #pragma unroll
  for (int t = 0; t < 4; ++t) {                  // C/D: VGPR i -> M=i+hf*8, N=r
    int n = n0 + t * 16 + r;
    float bv = bias[n];
    #pragma unroll
    for (int i = 0; i < 8; ++i) {
      int m = m0 + hf * 8 + i;
      float v = accs[t][i] + bv;
      if (out_mode == 0) {
        int bi = m / S, s = m % S;
        int hh = n / dk, c = n % dk;
        ((__bf16*)out)[(((size_t)bi * H + hh) * S + s) * dk + c] = f2bf(v);
      } else {
        ((float*)out)[(size_t)m * Ndim + n] = v;
      }
    }
  }
}

// ---------------------------------------------------------------------------
// Fused scores + decay + double softmax for a 16-row stripe of one (b,h).
// Q stripe (16x64 bf16, contiguous 2KB) staged into LDS by the TDM
// (tensor_load_to_lds, D# per ISA 08 §8), then every column tile's A-frag is
// a ds_load.  Phase 2 per row (1 wave/row): max -> sumE -> shuffle-scan
// cumsum of exp(s-mx) -> retention gate -> softmax-2 -> write attn.
// 8 waves / block. Dynamic LDS: 16*S floats (64 KB) + 2 KB static Q tile.
// ---------------------------------------------------------------------------
__global__ void __launch_bounds__(256)
attn_scores_softmax_kernel(const __bf16* __restrict__ Qb,
                           const __bf16* __restrict__ Kb,
                           const float* __restrict__ gammas,
                           float* __restrict__ attn, int S, int H) {
  extern __shared__ float sc[];                       // [16][S]
  __shared__ __bf16 qt[16 * 64];                      // staged Q stripe
  const int dk = 64;
  int lane = threadIdx.x & 31;
  int wave = threadIdx.x >> 5;
  int r = lane & 15, hf = lane >> 4;
  int tiles = S >> 4;
  int mt = blockIdx.x % tiles;
  int bh = blockIdx.x / tiles;
  int h  = bh % H;
  int m0 = mt << 4;
  float gamma = -fabsf(gammas[h]);
  const float scale = 0.125f;                          // 1/sqrt(64)

  // ----- stage Q stripe into LDS (TDM if available) -----
  const __bf16* qg = Qb + ((size_t)bh * S + m0) * dk;  // contiguous 16x64 tile
  if (wave == 0) {
#if defined(__gfx1250__) && __has_builtin(__builtin_amdgcn_tensor_load_to_lds)
    unsigned lds_off = (unsigned)(size_t)(void*)qt;    // LDS aperture low bits
    unsigned long long ga = (unsigned long long)(size_t)qg;
    // D# group0: count=1 | lds_addr | global_addr[56:0] | type=2
    v4u g0 = { 1u, lds_off, (unsigned)ga,
               (unsigned)((ga >> 32) & 0x01FFFFFFu) | (2u << 30) };
    // D# group1: data_size=2B; tensor_dim0=64, dim1=16; tile 64x16; stride0=64
    v8i g1 = { (int)0x00010000, (int)(64u << 16), (int)(16u << 16),
               (int)(64u << 16), 16, 64, 0, 0 };
    v4i gz4 = { 0, 0, 0, 0 };
    v8i gz8 = { 0, 0, 0, 0, 0, 0, 0, 0 };
    __builtin_amdgcn_tensor_load_to_lds(g0, g1, gz4, gz4, gz8, 0);
    __builtin_amdgcn_s_wait_tensorcnt(0);
#else
    for (int t = lane; t < 16 * 64; t += 32) qt[t] = qg[t];
#endif
  }
  __syncthreads();

  // ----- phase 1: score stripe via WMMA (causal: tiles <= mt) -----
  const __bf16* qrow = qt + r * dk;                    // A-frag from LDS
  for (int nt = wave; nt <= mt; nt += 8) {
    int n0 = nt << 4;
    const __bf16* krow = Kb + ((size_t)bh * S + n0 + r) * dk;
    v8f acc = {};
    #pragma unroll
    for (int k0 = 0; k0 < 64; k0 += 32) {
      v16bf a = load_a_frag(qrow, k0, hf);
      v16bf b = *(const v16bf*)(krow + k0 + hf * 16);
      acc = wmma_bf16(a, b, acc);
    }
    #pragma unroll
    for (int i = 0; i < 8; ++i)
      sc[(hf * 8 + i) * S + n0 + r] = acc[i] * scale;
  }
  __syncthreads();

  // ----- phase 2: rows (wave w handles rows w and w+8) -----
  for (int row = wave; row < 16; row += 8) {
    int qi = m0 + row;
    int L = qi + 1;                                    // causal valid length
    float* srow = sc + row * S;

    float mx = -3.0e38f;
    for (int j = lane; j < L; j += 32) mx = fmaxf(mx, srow[j]);
    mx = wave_max(mx);

    float se = 0.f;
    for (int j = lane; j < L; j += 32) se += __expf(srow[j] - mx);
    se = wave_sum(se);
    float inv = 1.0f / se;

    float running = 0.f, mx2 = -3.0e38f;
    for (int base = 0; base < L; base += 32) {
      int j = base + lane;
      float s = (j < L) ? srow[j] : 0.f;
      float e = (j < L) ? __expf(s - mx) : 0.f;
      float x = e;                                     // inclusive wave32 scan
      #pragma unroll
      for (int dd = 1; dd < 32; dd <<= 1) {
        float y = __shfl_up(x, dd, 32);
        if (lane >= dd) x += y;
      }
      float cum = (running + x) * inv;                 // distcum / disttot
      running += __shfl(x, 31, 32);
      if (j < L) {
        float rem  = fmaxf(1.0f - cum, 0.f);
        float pos  = (float)(qi - j);                  // |i-j|, j<=i
        float dist = sqrtf(rem * pos);
        float eff  = fminf(fmaxf(__expf(dist * gamma), 1e-5f), 1e5f);
        float g = s * eff;
        srow[j] = g;
        mx2 = fmaxf(mx2, g);
      }
    }
    mx2 = wave_max(mx2);

    float s2 = 0.f;
    for (int j = lane; j < L; j += 32) s2 += __expf(srow[j] - mx2);
    s2 = wave_sum(s2);
    float inv2 = 1.0f / s2;

    float* orow = attn + ((size_t)bh * S + qi) * S;    // masked tail -> 0
    for (int j = lane; j < S; j += 32)
      orow[j] = (j < L) ? __expf(srow[j] - mx2) * inv2 : 0.f;
  }
}

// ---------------------------------------------------------------------------
// O[b,s,h*dk+n] = sum_k attn[bh,m,k] * V[bh,k,n]  via WMMA.
// One wave = one 16-row M-tile x full dk: one A-frag (attn f32 -> bf16 on the
// fly, vector loads) feeds 4 accumulators. Causal K loop per wave; the
// over-read inside a 32-wide K chunk hits the exact zeros we wrote.
// attn stream (the 256 MB HBM re-read) is prefetched 2 chunks ahead.
// ---------------------------------------------------------------------------
__global__ void __launch_bounds__(128)
attn_v_kernel(const float* __restrict__ attn, const __bf16* __restrict__ Vb,
              __bf16* __restrict__ Obf, int S, int H) {
  const int dk = 64;
  int lane = threadIdx.x & 31;
  int wave = threadIdx.x >> 5;
  int r = lane & 15, hf = lane >> 4;
  int mblk = S >> 6;                                   // groups of 4 M-tiles
  int mt = (blockIdx.x % mblk) * 4 + wave;
  int bh = blockIdx.x / mblk;
  int b  = bh / H, h = bh % H;
  int m0 = mt << 4;
  int kend = (mt + 1) << 4;

  const float*  arow  = attn + ((size_t)bh * S + m0 + r) * S;
  const __bf16* vbase = Vb + (size_t)bh * S * dk;
  v8f acc[4] = {};
  for (int k0 = 0; k0 < kend; k0 += 32) {
    if (k0 + 64 < kend)                                // prefetch attn stream
      __builtin_prefetch((const void*)(arow + k0 + 64), 0, 3);
    v8f alo = *(const v8f*)(arow + k0 + hf * 16);
    v8f ahi = *(const v8f*)(arow + k0 + hf * 16 + 8);
    v16bf a;
    #pragma unroll
    for (int j = 0; j < 8; ++j) { a[j] = f2bf(alo[j]); a[j + 8] = f2bf(ahi[j]); }
    int kb = k0 + hf * 16;
    #pragma unroll
    for (int t = 0; t < 4; ++t) {
      v16bf bf;
      #pragma unroll
      for (int j = 0; j < 16; ++j)                     // B[k][n] = V[k][n]
        bf[j] = vbase[(size_t)(kb + j) * dk + t * 16 + r];
      acc[t] = wmma_bf16(a, bf, acc[t]);
    }
  }
  int dfull = H * dk;
  #pragma unroll
  for (int t = 0; t < 4; ++t)
    #pragma unroll
    for (int i = 0; i < 8; ++i) {
      int s = m0 + hf * 8 + i;
      Obf[((size_t)b * S + s) * dfull + h * dk + t * 16 + r] = f2bf(acc[t][i]);
    }
}

// ---------------------------------------------------------------------------
extern "C" void kernel_launch(void* const* d_in, const int* in_sizes, int n_in,
                              void* d_out, int out_size, void* d_ws, size_t ws_size,
                              hipStream_t stream) {
  (void)in_sizes; (void)n_in; (void)out_size; (void)ws_size;
  const int B = 4, S = 1024, d = 1024, H = 16, dk = 64;
  const int M = B * S;

  const float* query  = (const float*)d_in[0];
  const float* key_   = (const float*)d_in[1];
  const float* value  = (const float*)d_in[2];
  /* d_in[3] = causal mask, computed analytically */
  const float* Wq = (const float*)d_in[4];
  const float* bq = (const float*)d_in[5];
  const float* Wk = (const float*)d_in[6];
  const float* bk = (const float*)d_in[7];
  const float* Wv = (const float*)d_in[8];
  const float* bv = (const float*)d_in[9];
  const float* Wo = (const float*)d_in[10];
  const float* bo = (const float*)d_in[11];
  const float* gammas = (const float*)d_in[12];

  // workspace: 3*M*d + 4*d*d + 4*M*d bf16 = 32M elements = 64 MB
  __bf16* Xq  = (__bf16*)d_ws;
  __bf16* Xk  = Xq  + (size_t)M * d;
  __bf16* Xv  = Xk  + (size_t)M * d;
  __bf16* Wqb = Xv  + (size_t)M * d;
  __bf16* Wkb = Wqb + (size_t)d * d;
  __bf16* Wvb = Wkb + (size_t)d * d;
  __bf16* Wob = Wvb + (size_t)d * d;
  __bf16* Qb  = Wob + (size_t)d * d;     // [B,H,S,dk]
  __bf16* Kb  = Qb  + (size_t)M * d;
  __bf16* Vb  = Kb  + (size_t)M * d;
  __bf16* Ob  = Vb  + (size_t)M * d;     // [B,S,d]

  float* out  = (float*)d_out;
  float* attn = out + (size_t)M * d;     // [B,H,S,S], second tuple output

  const int nX = M * d, nW = d * d;
  f32_to_bf16_kernel<<<(nX / 4 + 255) / 256, 256, 0, stream>>>(query, Xq, nX);
  f32_to_bf16_kernel<<<(nX / 4 + 255) / 256, 256, 0, stream>>>(key_,  Xk, nX);
  f32_to_bf16_kernel<<<(nX / 4 + 255) / 256, 256, 0, stream>>>(value, Xv, nX);
  f32_to_bf16_kernel<<<(nW / 4 + 255) / 256, 256, 0, stream>>>(Wq, Wqb, nW);
  f32_to_bf16_kernel<<<(nW / 4 + 255) / 256, 256, 0, stream>>>(Wk, Wkb, nW);
  f32_to_bf16_kernel<<<(nW / 4 + 255) / 256, 256, 0, stream>>>(Wv, Wvb, nW);
  f32_to_bf16_kernel<<<(nW / 4 + 255) / 256, 256, 0, stream>>>(Wo, Wob, nW);

  dim3 gGrid(M / 16, d / 256), gBlk(128);
  gemm_bf16_kernel<<<gGrid, gBlk, 0, stream>>>(Xq, Wqb, bq, Qb, M, d, d, 0, S, H, dk);
  gemm_bf16_kernel<<<gGrid, gBlk, 0, stream>>>(Xk, Wkb, bk, Kb, M, d, d, 0, S, H, dk);
  gemm_bf16_kernel<<<gGrid, gBlk, 0, stream>>>(Xv, Wvb, bv, Vb, M, d, d, 0, S, H, dk);

  const int nblk = B * H * (S / 16);
  attn_scores_softmax_kernel<<<nblk, 256, 16 * S * sizeof(float), stream>>>(
      Qb, Kb, gammas, attn, S, H);

  attn_v_kernel<<<B * H * (S / 64), 128, 0, stream>>>(attn, Vb, Ob, S, H);

  gemm_bf16_kernel<<<gGrid, gBlk, 0, stream>>>(Ob, Wob, bo, out, M, d, d, 1, S, H, dk);
}
